// CustomRNN_69054484185642
// MI455X (gfx1250) — matrix operand order repaired and checked by reading, not compiled, not run on hardware
//
#include <hip/hip_runtime.h>
#include <hip/hip_bf16.h>

// ---------------------------------------------------------------------------
// 2-layer LSTM, B=64, T=256, N=M=1024.
//   1) convert x, Wx, Wh to bf16
//   2) pregate0[B*T,4096] = x @ Wx0^T + b0   (large bf16 WMMA GEMM)
//   3) per step: layer0 fused WMMA+cell, layer1 fused 2xWMMA+cell
//   4) final h/c states -> tail of d_out
// k-loops: 2-stage ping-pong pipeline enforced with sched_group_barrier
// (10 VMEM loads -> 4 WMMA -> 10 VMEM loads -> 4 WMMA per 64-wide k block),
// so global_load latency overlaps v_wmma with partial loadcnt waits.
// ---------------------------------------------------------------------------

#define BSZ   64
#define TLEN  256
#define NDIM  1024
#define MDIM  1024
#define GDIM  4096   // 4*M
#define KDIM  1024   // contraction dim (== NDIM == MDIM)

typedef __bf16 bf16_t;
typedef __attribute__((ext_vector_type(16))) __bf16 v16bf;
typedef __attribute__((ext_vector_type(8)))  __bf16 v8bf;
typedef __attribute__((ext_vector_type(8)))  float  v8f;

// sched_group_barrier masks
#define SG_WMMA      0x008   // MFMA/WMMA instructions
#define SG_VMEM_READ 0x020   // VMEM read instructions

// ---- WMMA fragment loaders (contiguous k order, same in A and B, wave32) ---------

// A: 16x32 bf16 tile, row-major source with leading dim lda.
// lane l (m=l&15, g=l>>4): elems 0..15 = K[g*16 .. g*16+15] (contiguous, matches B)
__device__ __forceinline__ v16bf load_fragA(const bf16_t* __restrict__ base,
                                            int lda, int lane) {
  const int m = lane & 15, g = lane >> 4;
  const bf16_t* p = base + (size_t)m * lda + g * 16;
  v8bf lo = *(const v8bf*)(p);
  v8bf hi = *(const v8bf*)(p + 8);
  v16bf a;
#pragma unroll
  for (int i = 0; i < 8; ++i) { a[i] = lo[i]; a[i + 8] = hi[i]; }
  return a;
}

// B: 32x16 bf16 tile where B[k][n] = W[n][k], W row-major with leading dim ldw.
// lane l (n=l&15, g=l>>4) holds K = g*16 + 0..15 (16 contiguous elems of row n).
__device__ __forceinline__ v16bf load_fragB(const bf16_t* __restrict__ Wrow0,
                                            int ldw, int lane) {
  const int n = lane & 15, g = lane >> 4;
  return *(const v16bf*)(Wrow0 + (size_t)n * ldw + g * 16);
}

__device__ __forceinline__ v8f wmma_bf16(v16bf a, v16bf b, v8f c) {
  return __builtin_amdgcn_wmma_f32_16x16x32_bf16(
      /*neg_a=*/false, a, /*neg_b=*/false, b,
      /*c_mod=*/(short)0, c, /*reuse_a=*/false, /*reuse_b=*/false);
}

__device__ __forceinline__ float sigm(float x) {
  return 1.0f / (1.0f + __expf(-x));
}

// ---- pipelined 16x64-tile GEMM accumulate over K=1024 ---------------------
// acc[q] += A(16x1024) x W[q]^T tile. 2-stage ping-pong; the schedule
// (loads of next stage before WMMAs of current stage) is pinned with
// sched_group_barrier so the backend cannot re-serialize it.
__device__ __forceinline__ void gemm_tile_acc(const bf16_t* __restrict__ Abase,
                                              const bf16_t* __restrict__ W0,
                                              const bf16_t* __restrict__ W1,
                                              const bf16_t* __restrict__ W2,
                                              const bf16_t* __restrict__ W3,
                                              v8f acc[4], int lane) {
  v16bf a0 = load_fragA(Abase, KDIM, lane);
  v16bf b00 = load_fragB(W0, KDIM, lane);
  v16bf b01 = load_fragB(W1, KDIM, lane);
  v16bf b02 = load_fragB(W2, KDIM, lane);
  v16bf b03 = load_fragB(W3, KDIM, lane);

#pragma unroll 1
  for (int k0 = 0; k0 < KDIM; k0 += 64) {
    // stage-1 prefetch (k0+32, always in range: KDIM % 64 == 0)
    const int k1 = k0 + 32;
    v16bf a1  = load_fragA(Abase + k1, KDIM, lane);
    v16bf b10 = load_fragB(W0 + k1, KDIM, lane);
    v16bf b11 = load_fragB(W1 + k1, KDIM, lane);
    v16bf b12 = load_fragB(W2 + k1, KDIM, lane);
    v16bf b13 = load_fragB(W3 + k1, KDIM, lane);

    acc[0] = wmma_bf16(a0, b00, acc[0]);
    acc[1] = wmma_bf16(a0, b01, acc[1]);
    acc[2] = wmma_bf16(a0, b02, acc[2]);
    acc[3] = wmma_bf16(a0, b03, acc[3]);

    // stage-0 prefetch ((k0+64) & 1023: harmless wrap on last iteration)
    const int k2 = (k0 + 64) & (KDIM - 1);
    a0  = load_fragA(Abase + k2, KDIM, lane);
    b00 = load_fragB(W0 + k2, KDIM, lane);
    b01 = load_fragB(W1 + k2, KDIM, lane);
    b02 = load_fragB(W2 + k2, KDIM, lane);
    b03 = load_fragB(W3 + k2, KDIM, lane);

    acc[0] = wmma_bf16(a1, b10, acc[0]);
    acc[1] = wmma_bf16(a1, b11, acc[1]);
    acc[2] = wmma_bf16(a1, b12, acc[2]);
    acc[3] = wmma_bf16(a1, b13, acc[3]);

    // Pin the pipelined schedule: loads(stage1) | wmma(stage0) |
    // loads(stage0') | wmma(stage1). 10 b128 loads per stage
    // (A: 2, B: 4x2), 4 WMMAs per stage.
    __builtin_amdgcn_sched_group_barrier(SG_VMEM_READ, 10, 0);
    __builtin_amdgcn_sched_group_barrier(SG_WMMA,       4, 0);
    __builtin_amdgcn_sched_group_barrier(SG_VMEM_READ, 10, 0);
    __builtin_amdgcn_sched_group_barrier(SG_WMMA,       4, 0);
  }
}

// ---- fp32 -> bf16 convert (grid-strided) ----------------------------------
__global__ void cvt_f32_bf16(const float* __restrict__ s, bf16_t* __restrict__ d,
                             int n) {
  int i = blockIdx.x * blockDim.x + threadIdx.x;
  int stride = gridDim.x * blockDim.x;
  for (; i < n; i += stride) d[i] = (bf16_t)s[i];
}

// ---- pregate0 = Xb @ Wx0^T + b0 -------------------------------------------
// Xb: [B*T, N] bf16, W: [4M, N] bf16, out: [B*T, 4M] f32.
__global__ __launch_bounds__(32, 1)
void gemm_pregate(const bf16_t* __restrict__ X, const bf16_t* __restrict__ W,
                  const float* __restrict__ bias, float* __restrict__ out) {
  const int lane = threadIdx.x;
  const int ct = blockIdx.x;   // 0..63  -> cols ct*64
  const int rt = blockIdx.y;   // 0..1023 -> rows rt*16
  const int n = lane & 15, g = lane >> 4;

  v8f acc[4];
#pragma unroll
  for (int q = 0; q < 4; ++q)
#pragma unroll
    for (int r = 0; r < 8; ++r) acc[q][r] = 0.0f;

  const bf16_t* Abase = X + (size_t)rt * 16 * NDIM;
  const bf16_t* Wb = W + (size_t)ct * 64 * NDIM;
  gemm_tile_acc(Abase, Wb, Wb + (size_t)16 * NDIM, Wb + (size_t)32 * NDIM,
                Wb + (size_t)48 * NDIM, acc, lane);

#pragma unroll
  for (int q = 0; q < 4; ++q)
#pragma unroll
    for (int r = 0; r < 8; ++r) {
      int row = rt * 16 + r + 8 * g;
      int col = ct * 64 + q * 16 + n;
      out[(size_t)row * GDIM + col] = acc[q][r] + bias[col];
    }
}

// ---- layer-0 step: gates = pregate(t) + h0 @ Wh0^T ; LSTM cell ------------
// grid (64,4): blockIdx.x = M col tile, blockIdx.y = batch tile. 1 wave/block.
__global__ __launch_bounds__(32, 1)
void lstm_step_l0(const float* __restrict__ pre,   // [B*T, 4M]
                  const bf16_t* __restrict__ Wh,   // [4M, M] bf16 (layer 0)
                  const bf16_t* __restrict__ h_in, // [B, M] bf16
                  bf16_t* __restrict__ h_out,      // [B, M] bf16
                  float* __restrict__ c_st,        // [B, M] f32 (in-place)
                  float* __restrict__ h_st,        // [B, M] f32
                  int t) {
  const int lane = threadIdx.x;
  const int mj = blockIdx.x;   // 0..63
  const int bi = blockIdx.y;   // 0..3
  const int n = lane & 15, g = lane >> 4;

  v8f acc[4];
#pragma unroll
  for (int q = 0; q < 4; ++q)
#pragma unroll
    for (int r = 0; r < 8; ++r) {
      int b = bi * 16 + r + 8 * g;
      acc[q][r] = pre[(size_t)(b * TLEN + t) * GDIM + q * MDIM + mj * 16 + n];
    }

  const bf16_t* Abase = h_in + (size_t)bi * 16 * MDIM;
  const bf16_t* Wb = Wh + (size_t)mj * 16 * MDIM;
  gemm_tile_acc(Abase, Wb, Wb + (size_t)MDIM * MDIM, Wb + (size_t)2 * MDIM * MDIM,
                Wb + (size_t)3 * MDIM * MDIM, acc, lane);

#pragma unroll
  for (int r = 0; r < 8; ++r) {
    int b = bi * 16 + r + 8 * g;
    int col = mj * 16 + n;
    size_t idx = (size_t)b * MDIM + col;
    float i_ = sigm(acc[0][r]);
    float f_ = sigm(acc[1][r]);
    float g_ = tanhf(acc[2][r]);
    float o_ = sigm(acc[3][r]);
    float cn = f_ * c_st[idx] + i_ * g_;
    float hn = o_ * tanhf(cn);
    c_st[idx] = cn;
    h_st[idx] = hn;
    h_out[idx] = (bf16_t)hn;
  }
}

// ---- layer-1 step: gates = b1 + h0 @ Wx1^T + h1 @ Wh1^T ; LSTM cell -------
// Two sequential pipelined GEMM passes sharing the accumulators.
__global__ __launch_bounds__(32, 1)
void lstm_step_l1(const float* __restrict__ bias,   // [4M] (layer 1)
                  const bf16_t* __restrict__ Wx,    // [4M, M] bf16 (layer 1)
                  const bf16_t* __restrict__ Wh,    // [4M, M] bf16 (layer 1)
                  const bf16_t* __restrict__ h0,    // [B, M] bf16 (this step's l0 out)
                  const bf16_t* __restrict__ h_in,  // [B, M] bf16 (l1 prev)
                  bf16_t* __restrict__ h_out,       // [B, M] bf16
                  float* __restrict__ c_st,         // [B, M] f32
                  float* __restrict__ h_st,         // [B, M] f32
                  float* __restrict__ outs,         // [B, T, M] f32 (d_out)
                  int t) {
  const int lane = threadIdx.x;
  const int mj = blockIdx.x;
  const int bi = blockIdx.y;
  const int n = lane & 15, g = lane >> 4;

  v8f acc[4];
#pragma unroll
  for (int q = 0; q < 4; ++q) {
    float bv = bias[q * MDIM + mj * 16 + n];
#pragma unroll
    for (int r = 0; r < 8; ++r) acc[q][r] = bv;
  }

  const bf16_t* A0 = h0 + (size_t)bi * 16 * MDIM;
  const bf16_t* Wxb = Wx + (size_t)mj * 16 * MDIM;
  gemm_tile_acc(A0, Wxb, Wxb + (size_t)MDIM * MDIM, Wxb + (size_t)2 * MDIM * MDIM,
                Wxb + (size_t)3 * MDIM * MDIM, acc, lane);

  const bf16_t* A1 = h_in + (size_t)bi * 16 * MDIM;
  const bf16_t* Whb = Wh + (size_t)mj * 16 * MDIM;
  gemm_tile_acc(A1, Whb, Whb + (size_t)MDIM * MDIM, Whb + (size_t)2 * MDIM * MDIM,
                Whb + (size_t)3 * MDIM * MDIM, acc, lane);

#pragma unroll
  for (int r = 0; r < 8; ++r) {
    int b = bi * 16 + r + 8 * g;
    int col = mj * 16 + n;
    size_t idx = (size_t)b * MDIM + col;
    float i_ = sigm(acc[0][r]);
    float f_ = sigm(acc[1][r]);
    float g_ = tanhf(acc[2][r]);
    float o_ = sigm(acc[3][r]);
    float cn = f_ * c_st[idx] + i_ * g_;
    float hn = o_ * tanhf(cn);
    c_st[idx] = cn;
    h_st[idx] = hn;
    h_out[idx] = (bf16_t)hn;
    outs[(size_t)b * TLEN * MDIM + (size_t)t * MDIM + col] = hn;
  }
}

// ---------------------------------------------------------------------------
extern "C" void kernel_launch(void* const* d_in, const int* in_sizes, int n_in,
                              void* d_out, int out_size, void* d_ws, size_t ws_size,
                              hipStream_t stream) {
  const float* x  = (const float*)d_in[0];  // [B,T,N]
  const float* h0 = (const float*)d_in[1];  // [L,B,M]
  const float* c0 = (const float*)d_in[2];  // [L,B,M]
  const float* Wx = (const float*)d_in[3];  // [L,4M,N]
  const float* Wh = (const float*)d_in[4];  // [L,4M,M]
  const float* bb = (const float*)d_in[5];  // [L,4M]
  float* out = (float*)d_out;               // outs | h_last | c_last

  const size_t nX  = (size_t)BSZ * TLEN * NDIM;   // 16,777,216
  const size_t nW  = (size_t)2 * GDIM * NDIM;     //  8,388,608 per weight set
  const size_t nBM = (size_t)BSZ * MDIM;          //     65,536
  const size_t nPre = (size_t)BSZ * TLEN * GDIM;  // 67,108,864

  // workspace carve-up (~337 MB total; assumed <= ws_size)
  char* w = (char*)d_ws;
  bf16_t* xb   = (bf16_t*)w;  w += nX * 2;
  bf16_t* wxb  = (bf16_t*)w;  w += nW * 2;
  bf16_t* whb  = (bf16_t*)w;  w += nW * 2;
  float*  pre  = (float*)w;   w += nPre * 4;
  float*  h_st = (float*)w;   w += 2 * nBM * 4;   // [L,B,M]
  float*  c_st = (float*)w;   w += 2 * nBM * 4;   // [L,B,M]
  bf16_t* h0b[2]; h0b[0] = (bf16_t*)w; w += nBM * 2; h0b[1] = (bf16_t*)w; w += nBM * 2;
  bf16_t* h1b[2]; h1b[0] = (bf16_t*)w; w += nBM * 2; h1b[1] = (bf16_t*)w; w += nBM * 2;

  // 1) bf16 conversions + state init
  cvt_f32_bf16<<<4096, 256, 0, stream>>>(x,  xb,  (int)nX);
  cvt_f32_bf16<<<4096, 256, 0, stream>>>(Wx, wxb, (int)nW);
  cvt_f32_bf16<<<4096, 256, 0, stream>>>(Wh, whb, (int)nW);
  cvt_f32_bf16<<<256, 256, 0, stream>>>(h0,        h0b[0], (int)nBM);
  cvt_f32_bf16<<<256, 256, 0, stream>>>(h0 + nBM,  h1b[0], (int)nBM);
  hipMemcpyAsync(h_st, h0, 2 * nBM * sizeof(float), hipMemcpyDeviceToDevice, stream);
  hipMemcpyAsync(c_st, c0, 2 * nBM * sizeof(float), hipMemcpyDeviceToDevice, stream);

  // 2) precompute layer-0 input gates for all timesteps (the big WMMA GEMM)
  gemm_pregate<<<dim3(GDIM / 64, (BSZ * TLEN) / 16), 32, 0, stream>>>(
      xb, wxb /* layer0 Wx */, bb /* layer0 bias */, pre);

  // 3) sequential recurrence (graph-capture friendly: plain stream enqueues)
  const bf16_t* wx1 = wxb + (size_t)GDIM * NDIM;  // layer1 Wx bf16
  const bf16_t* wh0 = whb;                        // layer0 Wh bf16
  const bf16_t* wh1 = whb + (size_t)GDIM * MDIM;  // layer1 Wh bf16
  for (int t = 0; t < TLEN; ++t) {
    int pi = t & 1, po = (t + 1) & 1;
    lstm_step_l0<<<dim3(MDIM / 16, BSZ / 16), 32, 0, stream>>>(
        pre, wh0, h0b[pi], h0b[po], c_st, h_st, t);
    lstm_step_l1<<<dim3(MDIM / 16, BSZ / 16), 32, 0, stream>>>(
        bb + GDIM, wx1, wh1, h0b[po], h1b[pi], h1b[po],
        c_st + nBM, h_st + nBM, out, t);
  }

  // 4) final h/c states -> tail of d_out
  float* h_last = out + (size_t)BSZ * TLEN * MDIM;
  float* c_last = h_last + 2 * nBM;
  hipMemcpyAsync(h_last, h_st, 2 * nBM * sizeof(float), hipMemcpyDeviceToDevice, stream);
  hipMemcpyAsync(c_last, c_st, 2 * nBM * sizeof(float), hipMemcpyDeviceToDevice, stream);
}
